// URA_7155415515737
// MI455X (gfx1250) — matrix operand order repaired and checked by reading, not compiled, not run on hardware
//
#include <hip/hip_runtime.h>

typedef _Float16 h16;
typedef __attribute__((ext_vector_type(16))) _Float16 v16h;
typedef __attribute__((ext_vector_type(8)))  float    v8f;
typedef __attribute__((ext_vector_type(4)))  unsigned v4u;
typedef __attribute__((ext_vector_type(8)))  unsigned v8u;

#define B_  32
#define C_  64
#define H_  48
#define W_  48
#define N_  2304   // H*W
#define HL_ 24
#define WL_ 24
#define M_  576    // HL*WL

// workspace byte offsets (total ~50.2 MB)
#define OFF_CGN   0u          // B*N f32 (pre-blur cg, later normalized cgN)
#define OFF_BLUR  294912u     // B*N f32
#define OFF_CGM   589824u     // B*M f32
#define OFF_MAX   663552u     // 1 int (float bits), padded
#define OFF_Q     663808u     // B*N*64 f16
#define OFF_K     10100992u   // B*M*64 f16
#define OFF_V     12460288u   // B*M*64 f16
#define OFF_ATT   14819584u   // B*64*N f32 (attn map, NCHW)
#define OFF_T     33693952u   // B*64*N f32 (x + relu(conv1))

// gaussian kernel, sigma=1, size 7, normalized
__constant__ float G7[7] = {
    0.004433048175f, 0.054005582622f, 0.242036229376f, 0.399050279652f,
    0.242036229376f, 0.054005582622f, 0.004433048175f
};

__device__ __forceinline__ v8f wmma_f16(v16h a, v16h b, v8f c) {
    return __builtin_amdgcn_wmma_f32_16x16x32_f16(
        /*neg_a=*/false, a, /*neg_b=*/false, b,
        /*c_mod=*/(short)0, c, /*reuse_a=*/false, /*reuse_b=*/false);
}

// Tensor Data Mover: DMA a 2-D f16 panel [rows x 16] (row stride 64 elements)
// from global memory into LDS at lds_off. D# per CDNA5 ISA.
__device__ __forceinline__ void tdm_load_panel_f16(const h16* gptr, unsigned lds_off) {
    unsigned long long ga = (unsigned long long)(uintptr_t)gptr;
    v4u g0;
    g0[0] = 1u;                                   // count=1 (valid user descriptor)
    g0[1] = lds_off;                              // lds_addr (bytes)
    g0[2] = (unsigned)(ga & 0xFFFFFFFFu);         // global_addr[31:0]
    g0[3] = (unsigned)((ga >> 32) & 0x01FFFFFFu)  // global_addr[56:32]
            | (2u << 30);                         // type=2 ("image")
    v8u g1;
    g1[0] = 1u << 16;                             // data_size=1 -> 2 bytes
    g1[1] = (unsigned)C_ << 16;                   // tensor_dim0 = 64 (low16)
    g1[2] = (unsigned)M_ << 16;                   // tensor_dim1 = 576 (low16)
    g1[3] = 16u << 16;                            // tile_dim0 = 16
    g1[4] = (unsigned)M_;                         // tile_dim1 = 576, tile_dim2 = 0
    g1[5] = (unsigned)C_;                         // tensor_dim0_stride = 64 (low32)
    g1[6] = 0u;
    g1[7] = 0u;
    asm volatile("tensor_load_to_lds %0, %1" :: "s"(g0), "s"(g1) : "memory");
}

// ---------------- Stage 1: uncertainty map ----------------

__global__ void k_uncertain(const float* __restrict__ ms, float* __restrict__ cg0) {
    int idx = blockIdx.x * blockDim.x + threadIdx.x;
    if (idx >= B_ * N_) return;
    int b = idx / N_, p = idx % N_;
    int y = p / W_, x = p % W_;
    // half-pixel bilinear, 24 -> 48
    float sy = 0.5f * y - 0.25f, sx = 0.5f * x - 0.25f;
    float fy = floorf(sy), fx = floorf(sx);
    float wy = sy - fy, wx = sx - fx;
    int y0 = min(HL_ - 1, max(0, (int)fy));
    int y1 = min(HL_ - 1, max(0, (int)fy + 1));
    int x0 = min(WL_ - 1, max(0, (int)fx));
    int x1 = min(WL_ - 1, max(0, (int)fx + 1));
    const float* mb = ms + b * (HL_ * WL_);
    float v00 = mb[y0 * WL_ + x0], v01 = mb[y0 * WL_ + x1];
    float v10 = mb[y1 * WL_ + x0], v11 = mb[y1 * WL_ + x1];
    float v0 = v00 + (v01 - v00) * wx;
    float v1 = v10 + (v11 - v10) * wx;
    float v  = v0 + (v1 - v0) * wy;
    float s = 1.0f / (1.0f + __expf(-v));
    cg0[idx] = 0.5f - fabsf(s - 0.5f);
}

__device__ __forceinline__ int refl48(int i) {
    return i < 0 ? -i : (i >= H_ ? 2 * H_ - 2 - i : i);
}

__global__ void k_blur(const float* __restrict__ cg0, float* __restrict__ cgb,
                       int* __restrict__ maxbits) {
    int idx = blockIdx.x * blockDim.x + threadIdx.x;
    if (idx >= B_ * N_) return;
    int b = idx / N_, p = idx % N_;
    int y = p / W_, x = p % W_;
    const float* sb = cg0 + b * N_;
    float acc = 0.f;
    #pragma unroll
    for (int dy = 0; dy < 7; ++dy) {
        int yy = refl48(y + dy - 3);
        float racc = 0.f;
        #pragma unroll
        for (int dx = 0; dx < 7; ++dx) {
            int xx = refl48(x + dx - 3);
            racc += G7[dx] * sb[yy * W_ + xx];
        }
        acc += G7[dy] * racc;
    }
    acc *= 4.0f;
    cgb[idx] = acc;
    atomicMax(maxbits, __float_as_int(acc));  // values are non-negative
}

__global__ void k_norm(const float* __restrict__ cgb, const int* __restrict__ maxbits,
                       float* __restrict__ cgN, float* __restrict__ cgM,
                       float* __restrict__ outCg) {
    int idx = blockIdx.x * blockDim.x + threadIdx.x;
    if (idx >= B_ * N_) return;
    float inv = 1.0f / __int_as_float(*maxbits);
    float v = cgb[idx] * inv;
    cgN[idx]  = v;
    outCg[idx] = v;
    int b = idx / N_, p = idx % N_;
    int y = p / W_, x = p % W_;
    if (((y | x) & 1) == 0)  // nearest down: src = 2*dst
        cgM[b * M_ + (y >> 1) * WL_ + (x >> 1)] = v;
}

// ---------------- Stage 2: WMMA projections ----------------

__global__ __launch_bounds__(32)
void k_proj_q(const float* __restrict__ x, const float* __restrict__ Wq,
              const float* __restrict__ bq, h16* __restrict__ Qf) {
    int t = blockIdx.x;
    int b = t / (144 * 4);
    int rem = t % (144 * 4);
    int nt = rem >> 2, ct = rem & 3;
    int lane = threadIdx.x;
    int lo = lane & 15, hi = lane >> 4;
    int n = nt * 16 + lo;           // A-row this lane owns
    int cout = ct * 16 + lo;        // B/C column this lane owns
    const float* xb = x + (size_t)b * C_ * N_;
    v8f acc;
    float bias = bq[cout];
    #pragma unroll
    for (int i = 0; i < 8; ++i) acc[i] = bias;
    #pragma unroll
    for (int s = 0; s < 2; ++s) {
        v16h a, bb;
        int ab = s * 32 + hi * 16;
        #pragma unroll
        for (int i = 0; i < 8; ++i) a[i]     = (h16)xb[(size_t)(ab + i) * N_ + n];
        #pragma unroll
        for (int i = 0; i < 8; ++i) a[8 + i] = (h16)xb[(size_t)(ab + 8 + i) * N_ + n];
        const float* wr = Wq + cout * C_ + s * 32 + hi * 16;
        #pragma unroll
        for (int i = 0; i < 16; ++i) bb[i] = (h16)wr[i];
        acc = wmma_f16(a, bb, acc);
    }
    h16* qb = Qf + (size_t)b * N_ * C_;
    #pragma unroll
    for (int r = 0; r < 8; ++r) {
        int nr = nt * 16 + r + hi * 8;
        qb[(size_t)nr * C_ + cout] = (h16)acc[r];
    }
}

__global__ __launch_bounds__(32)
void k_proj_kv(const float* __restrict__ l,
               const float* __restrict__ Wk, const float* __restrict__ bk,
               const float* __restrict__ Wv, const float* __restrict__ bv,
               h16* __restrict__ Kf, h16* __restrict__ Vf) {
    int t = blockIdx.x;
    int b = t / (36 * 4);
    int rem = t % (36 * 4);
    int mt = rem >> 2, ct = rem & 3;
    int lane = threadIdx.x;
    int lo = lane & 15, hi = lane >> 4;
    int m = mt * 16 + lo;
    int cout = ct * 16 + lo;
    const float* lb = l + (size_t)b * C_ * M_;
    v8f accK, accV;
    float biask = bk[cout], biasv = bv[cout];
    #pragma unroll
    for (int i = 0; i < 8; ++i) { accK[i] = biask; accV[i] = biasv; }
    #pragma unroll
    for (int s = 0; s < 2; ++s) {
        v16h a, bbk, bbv;
        int ab = s * 32 + hi * 16;
        #pragma unroll
        for (int i = 0; i < 8; ++i) a[i]     = (h16)lb[(size_t)(ab + i) * M_ + m];
        #pragma unroll
        for (int i = 0; i < 8; ++i) a[8 + i] = (h16)lb[(size_t)(ab + 8 + i) * M_ + m];
        const float* wrk = Wk + cout * C_ + s * 32 + hi * 16;
        const float* wrv = Wv + cout * C_ + s * 32 + hi * 16;
        #pragma unroll
        for (int i = 0; i < 16; ++i) { bbk[i] = (h16)wrk[i]; bbv[i] = (h16)wrv[i]; }
        accK = wmma_f16(a, bbk, accK);
        accV = wmma_f16(a, bbv, accV);
    }
    #pragma unroll
    for (int r = 0; r < 8; ++r) {
        int mr = mt * 16 + r + hi * 8;
        Kf[((size_t)b * M_ + mr) * C_ + cout] = (h16)accK[r];
        Vf[((size_t)b * M_ + mr) * C_ + cout] = (h16)accV[r];
    }
}

// ---------------- Stage 3: fused attention (one wave per 16-row tile) ----------------
// V panels are DMAed into LDS by the Tensor Data Mover, overlapped with QK^T + softmax.

__global__ __launch_bounds__(32)
void k_attn(const h16* __restrict__ Qf, const h16* __restrict__ Kf,
            const h16* __restrict__ Vf, const float* __restrict__ cgN,
            const float* __restrict__ cgM, float* __restrict__ attn) {
    __shared__ float S[16 * M_];      // 36 KB score block
    __shared__ h16   Vsh[M_ * 16];    // 18 KB V panel (TDM destination)
    __shared__ float cgr[16];
    int t = blockIdx.x;
    int b = t / 144, nt = t % 144;
    int lane = threadIdx.x;
    int lo = lane & 15, hi = lane >> 4;

    const h16* kb_ = Kf + (size_t)b * M_ * C_;
    const h16* vb_ = Vf + (size_t)b * M_ * C_;
    // generic shared pointer: LDS offset lives in the low 32 bits (aperture in [63:32])
    unsigned vsh_off = (unsigned)(uintptr_t)(&Vsh[0]);

    // kick off DMA of V panel ct=0 while we compute QK^T and softmax
    tdm_load_panel_f16(vb_, vsh_off);

    if (lane < 16) cgr[lane] = cgN[b * N_ + nt * 16 + lane];
    __syncthreads();

    // A fragments: Q rows, K-contraction over C=64 (two K=32 steps)
    const h16* qrow = Qf + ((size_t)b * N_ + nt * 16 + lo) * C_;
    v16h a0, a1;
    #pragma unroll
    for (int i = 0; i < 8; ++i) {
        a0[i]     = qrow[hi * 16 + i];
        a0[8 + i] = qrow[hi * 16 + 8 + i];
        a1[i]     = qrow[32 + hi * 16 + i];
        a1[8 + i] = qrow[32 + hi * 16 + 8 + i];
    }

    // S = Q K^T * scale + log(cgN cgM + 0.5)
    for (int mt = 0; mt < 36; ++mt) {
        int m = mt * 16 + lo;
        const h16* krow = kb_ + (size_t)m * C_;
        if (mt + 1 < 36)
            __builtin_prefetch(kb_ + (size_t)(m + 16) * C_, 0, 1);
        v16h b0, b1;
        #pragma unroll
        for (int i = 0; i < 16; ++i) {
            b0[i] = krow[hi * 16 + i];
            b1[i] = krow[32 + hi * 16 + i];
        }
        v8f acc;
        #pragma unroll
        for (int i = 0; i < 8; ++i) acc[i] = 0.f;
        acc = wmma_f16(a0, b0, acc);
        acc = wmma_f16(a1, b1, acc);
        float cm = cgM[b * M_ + m];
        #pragma unroll
        for (int r = 0; r < 8; ++r) {
            int rr = r + hi * 8;
            S[rr * M_ + m] = acc[r] * 0.125f + __logf(cgr[rr] * cm + 0.5f);
        }
    }
    __syncthreads();

    // row softmax, normalized in place (P stored in LDS)
    if (lane < 16) {
        float* srow = S + lane * M_;
        float mx = -3.4e38f;
        for (int j = 0; j < M_; ++j) mx = fmaxf(mx, srow[j]);
        float sum = 0.f;
        for (int j = 0; j < M_; ++j) { float e = __expf(srow[j] - mx); srow[j] = e; sum += e; }
        float inv = 1.0f / sum;
        for (int j = 0; j < M_; ++j) srow[j] *= inv;
    }
    __syncthreads();

    // O = P V  (K-contraction over M=576: 18 steps of K=32, 4 column tiles)
    // B fragments come from the TDM-staged LDS panel.
    float* ab = attn + (size_t)b * C_ * N_;
    const float* srow = S + lo * M_;
    for (int ct = 0; ct < 4; ++ct) {
        int cc = ct * 16 + lo;
        __builtin_amdgcn_s_wait_tensorcnt(0);   // panel ct resident in LDS
        __syncthreads();
        v8f acc;
        #pragma unroll
        for (int i = 0; i < 8; ++i) acc[i] = 0.f;
        for (int kt = 0; kt < 18; ++kt) {
            v16h a, bb;
            int ab0 = kt * 32 + hi * 16;
            #pragma unroll
            for (int i = 0; i < 8; ++i) {
                a[i]     = (h16)srow[ab0 + i];
                a[8 + i] = (h16)srow[ab0 + 8 + i];
            }
            int kb0 = kt * 32 + hi * 16;
            #pragma unroll
            for (int i = 0; i < 16; ++i)
                bb[i] = Vsh[(kb0 + i) * 16 + lo];
            acc = wmma_f16(a, bb, acc);
        }
        if (ct < 3) {
            // panel fully consumed (LDS reads landed before the WMMAs); start next DMA
            asm volatile("s_wait_dscnt 0x0" ::: "memory");
            tdm_load_panel_f16(vb_ + (size_t)(ct + 1) * 16, vsh_off);
        }
        #pragma unroll
        for (int r = 0; r < 8; ++r) {
            int n = nt * 16 + r + hi * 8;
            ab[(size_t)cc * N_ + n] = acc[r];
        }
    }
}

// ---------------- Stage 4: convolutions ----------------

// out = relu(conv3x3(in, w) + bias) [+ resid]; one block per (b, co), weights in LDS
__global__ __launch_bounds__(256)
void k_conv3x3(const float* __restrict__ in, const float* __restrict__ w,
               const float* __restrict__ bias, const float* __restrict__ resid,
               float* __restrict__ out) {
    __shared__ float wsh[C_ * 9];
    int blk = blockIdx.x;
    int b = blk >> 6, co = blk & 63;
    for (int i = threadIdx.x; i < C_ * 9; i += blockDim.x)
        wsh[i] = w[co * C_ * 9 + i];
    __syncthreads();
    float bsv = bias[co];
    const float* inb = in + (size_t)b * C_ * N_;
    size_t obase = ((size_t)b * C_ + co) * N_;
    for (int p = threadIdx.x; p < N_; p += blockDim.x) {
        int y = p / W_, x = p % W_;
        float acc = bsv;
        for (int ci = 0; ci < C_; ++ci) {
            const float* ip = inb + (size_t)ci * N_;
            const float* wp = wsh + ci * 9;
            #pragma unroll
            for (int dy = -1; dy <= 1; ++dy) {
                int yy = y + dy;
                if (yy < 0 || yy >= H_) continue;
                #pragma unroll
                for (int dx = -1; dx <= 1; ++dx) {
                    int xx = x + dx;
                    if (xx < 0 || xx >= W_) continue;
                    acc += wp[(dy + 1) * 3 + (dx + 1)] * ip[yy * W_ + xx];
                }
            }
        }
        acc = fmaxf(acc, 0.f);
        if (resid) acc += resid[obase + p];
        out[obase + p] = acc;
    }
}

__global__ __launch_bounds__(256)
void k_conv1x1(const float* __restrict__ X2, const float* __restrict__ w4,
               const float* __restrict__ b4, float* __restrict__ out) {
    __shared__ float wsh[C_];
    if (threadIdx.x < C_) wsh[threadIdx.x] = w4[threadIdx.x];
    __syncthreads();
    int idx = blockIdx.x * blockDim.x + threadIdx.x;
    if (idx >= B_ * N_) return;
    int b = idx / N_, p = idx % N_;
    const float* xb = X2 + (size_t)b * C_ * N_;
    float acc = b4[0];
    #pragma unroll
    for (int c = 0; c < C_; ++c) acc += wsh[c] * xb[(size_t)c * N_ + p];
    out[idx] = acc;
}

// ---------------- launcher ----------------

extern "C" void kernel_launch(void* const* d_in, const int* in_sizes, int n_in,
                              void* d_out, int out_size, void* d_ws, size_t ws_size,
                              hipStream_t stream) {
    const float* x   = (const float*)d_in[0];
    const float* l   = (const float*)d_in[1];
    const float* ms  = (const float*)d_in[2];
    const float* Wq  = (const float*)d_in[3];
    const float* bq  = (const float*)d_in[4];
    const float* Wk  = (const float*)d_in[5];
    const float* bk  = (const float*)d_in[6];
    const float* Wv  = (const float*)d_in[7];
    const float* bv  = (const float*)d_in[8];
    const float* w1  = (const float*)d_in[9];
    const float* b1  = (const float*)d_in[10];
    const float* w3  = (const float*)d_in[11];
    const float* b3  = (const float*)d_in[12];
    const float* w4  = (const float*)d_in[13];
    const float* b4  = (const float*)d_in[14];

    float* out = (float*)d_out;
    char*  ws  = (char*)d_ws;
    float* cgN   = (float*)(ws + OFF_CGN);
    float* cgB   = (float*)(ws + OFF_BLUR);
    float* cgM   = (float*)(ws + OFF_CGM);
    int*   maxb  = (int*)  (ws + OFF_MAX);
    h16*   Qf    = (h16*)  (ws + OFF_Q);
    h16*   Kf    = (h16*)  (ws + OFF_K);
    h16*   Vf    = (h16*)  (ws + OFF_V);
    float* attnB = (float*)(ws + OFF_ATT);
    float* Tbuf  = (float*)(ws + OFF_T);

    float* outX  = out;                          // [B,C,H,W]
    float* outO  = out + (size_t)B_ * C_ * N_;   // [B,1,H,W]
    float* outCg = outO + (size_t)B_ * N_;       // [B,1,H,W]

    hipMemsetAsync(maxb, 0, sizeof(int), stream);

    const int tpb = 256;
    const int nPix = B_ * N_;
    const int gPix = (nPix + tpb - 1) / tpb;

    k_uncertain<<<gPix, tpb, 0, stream>>>(ms, cgN);
    k_blur<<<gPix, tpb, 0, stream>>>(cgN, cgB, maxb);
    k_norm<<<gPix, tpb, 0, stream>>>(cgB, maxb, cgN, cgM, outCg);

    k_proj_q <<<B_ * 144 * 4, 32, 0, stream>>>(x, Wq, bq, Qf);
    k_proj_kv<<<B_ * 36 * 4,  32, 0, stream>>>(l, Wk, bk, Wv, bv, Kf, Vf);
    k_attn   <<<B_ * 144,     32, 0, stream>>>(Qf, Kf, Vf, cgN, cgM, attnB);

    k_conv3x3<<<B_ * C_, tpb, 0, stream>>>(attnB, w1, b1, x, Tbuf);
    k_conv3x3<<<B_ * C_, tpb, 0, stream>>>(Tbuf, w3, b3, nullptr, outX);
    k_conv1x1<<<gPix, tpb, 0, stream>>>(outX, w4, b4, outO);
}
